// Net_MP_46849503265410
// MI455X (gfx1250) — matrix operand order repaired and checked by reading, not compiled, not run on hardware
//
#include <hip/hip_runtime.h>

typedef __attribute__((ext_vector_type(16))) _Float16 v16h;
typedef __attribute__((ext_vector_type(8)))  float    v8f;

#define NN 100000
#define NE 200000

// ---------------------------------------------------------------------------
// prep: swizzle concatenated edge-MLP second-layer weights into WMMA B-fragment
// order (f16). Wcat2[544][32]: rows k*32+i = e2_w2[k][i*32+o], rows 512+i = e2_b2.
// Wcat1[64][32]: rows k*3+i = e1_w2[k][i*32+o] (48 rows), rows 48..50 = e1_b2, pad 0.
// Stored as frag[(c*2+nt)*32 + lane][t] = Wcat[c*32 + (t&8)*2 + (lane>>4)*8 + (t&7)][nt*16 + (lane&15)]
// ---------------------------------------------------------------------------
__global__ void prep_kernel(const float* __restrict__ e1_w2, const float* __restrict__ e1_b2,
                            const float* __restrict__ e2_w2, const float* __restrict__ e2_b2,
                            _Float16* __restrict__ WB2, _Float16* __restrict__ WB1)
{
    int gid = blockIdx.x * blockDim.x + threadIdx.x;
    if (gid < 17 * 1024) {                       // WB2: 17 chunks * 2 ntiles * 32 lanes * 16
        int c    = gid >> 10;
        int rem  = gid & 1023;
        int nt   = rem >> 9;
        int lane = (rem >> 4) & 31;
        int t    = rem & 15;
        int i    = (t < 8) ? ((lane >> 4) * 8 + t) : (16 + (lane >> 4) * 8 + (t - 8));       // K within chunk = i (inner dim is 32)
        int col  = nt * 16 + (lane & 15);
        float v  = (c < 16) ? e2_w2[c * 1024 + i * 32 + col] : e2_b2[i * 32 + col];
        WB2[gid] = (_Float16)v;
    } else if (gid < 17 * 1024 + 2048) {         // WB1: 2 chunks * 2 ntiles * 32 lanes * 16
        int idx  = gid - 17 * 1024;
        int c    = idx >> 10;
        int rem  = idx & 1023;
        int nt   = rem >> 9;
        int lane = (rem >> 4) & 31;
        int t    = rem & 15;
        int kr   = c * 32 + ((t < 8) ? ((lane >> 4) * 8 + t) : (16 + (lane >> 4) * 8 + (t - 8)));
        int col  = nt * 16 + (lane & 15);
        float v  = 0.f;
        if (kr < 48)      v = e1_w2[(kr / 3) * 96 + (kr % 3) * 32 + col];
        else if (kr < 51) v = e1_b2[(kr - 48) * 32 + col];
        WB1[idx] = (_Float16)v;
    }
}

// ---------------------------------------------------------------------------
// edge_g: per-edge first MLP layers  g = relu(ea @ w1 + b1) for both convs,
// stored f16; also accumulate in-degree counts.
// ---------------------------------------------------------------------------
__global__ void edge_g_kernel(const float* __restrict__ ea, const int* __restrict__ dst,
                              const float* __restrict__ w1a, const float* __restrict__ b1a,
                              const float* __restrict__ w1b, const float* __restrict__ b1b,
                              _Float16* __restrict__ g1h, _Float16* __restrict__ g2h,
                              float* __restrict__ cnt, int E)
{
    int e = blockIdx.x * blockDim.x + threadIdx.x;
    if (e >= E) return;
    float a0 = ea[e * 4 + 0], a1 = ea[e * 4 + 1], a2 = ea[e * 4 + 2], a3 = ea[e * 4 + 3];
    v16h g1, g2;
#pragma unroll
    for (int j = 0; j < 16; ++j) {
        float t1 = b1a[j] + a0 * w1a[j] + a1 * w1a[16 + j] + a2 * w1a[32 + j] + a3 * w1a[48 + j];
        float t2 = b1b[j] + a0 * w1b[j] + a1 * w1b[16 + j] + a2 * w1b[32 + j] + a3 * w1b[48 + j];
        g1[j] = (_Float16)fmaxf(t1, 0.f);
        g2[j] = (_Float16)fmaxf(t2, 0.f);
    }
    *(v16h*)(g1h + (size_t)e * 16) = g1;
    *(v16h*)(g2h + (size_t)e * 16) = g2;
    atomicAdd(cnt + dst[e], 1.0f);
}

// ---------------------------------------------------------------------------
// Layer-1 A-fragment builder. P1[e][q]: q=k*3+i (q<48) -> g[k]*x[i];
// q=48..50 -> x[q-48]; q>=51 -> 0.  KBASE = (lane>>4)*8, compile-time per branch
// so all gr[] indices are constants after unroll.
// ---------------------------------------------------------------------------
template <int KBASE>
__device__ __forceinline__ void build_a1(const v16h& gr, float x0, float x1, float x2,
                                         v16h& a0, v16h& a1)
{
#pragma unroll
    for (int t = 0; t < 16; ++t) {
        const int q0 = (t < 8) ? (KBASE + t) : (16 + KBASE + (t - 8));   // <= 31, always < 48
        const int q1 = 32 + q0;                                          // 32..63
        float v0 = (float)gr[q0 / 3] * ((q0 % 3 == 0) ? x0 : ((q0 % 3 == 1) ? x1 : x2));
        float v1;
        if (q1 < 48)      v1 = (float)gr[q1 / 3] * ((q1 % 3 == 0) ? x0 : ((q1 % 3 == 1) ? x1 : x2));
        else if (q1 < 51) v1 = (q1 == 48) ? x0 : ((q1 == 49) ? x1 : x2);
        else              v1 = 0.f;
        a0[t] = (_Float16)v0;
        a1[t] = (_Float16)v1;
    }
}

// ---------------------------------------------------------------------------
// edge_msg1: one wave per 16-edge tile. msg1 = P1[16 x 64] @ Wcat1[64 x 32]
// via 4x v_wmma_f32_16x16x32_f16, scatter-add into acc1.
// ---------------------------------------------------------------------------
__global__ void edge_msg1_kernel(const int* __restrict__ src, const int* __restrict__ dst,
                                 const _Float16* __restrict__ g1h, const float* __restrict__ x,
                                 const _Float16* __restrict__ WB1, float* __restrict__ acc1, int E)
{
    int lane = threadIdx.x & 31;
    int wave = threadIdx.x >> 5;
    int tile = blockIdx.x * (blockDim.x >> 5) + wave;
    int base = tile * 16;
    if (base >= E) return;               // wave-uniform exit; EXEC all-1s for WMMA
    int row = lane & 15;
    int half = lane >> 4;
    int e = base + row;
    int s = src[e];
    int dall = dst[e];
    float x0 = x[s * 3 + 0], x1 = x[s * 3 + 1], x2 = x[s * 3 + 2];
    v16h gr = *(const v16h*)(g1h + (size_t)e * 16);
    v16h a0, a1;
    if (half == 0) build_a1<0>(gr, x0, x1, x2, a0, a1);
    else           build_a1<8>(gr, x0, x1, x2, a0, a1);
    const v16h* wb = (const v16h*)WB1;
    v8f c0 = {}, c1 = {};
    c0 = __builtin_amdgcn_wmma_f32_16x16x32_f16(false, a0, false, wb[0 * 32 + lane], (short)0, c0, false, false);
    c1 = __builtin_amdgcn_wmma_f32_16x16x32_f16(false, a0, false, wb[1 * 32 + lane], (short)0, c1, false, false);
    c0 = __builtin_amdgcn_wmma_f32_16x16x32_f16(false, a1, false, wb[2 * 32 + lane], (short)0, c0, false, false);
    c1 = __builtin_amdgcn_wmma_f32_16x16x32_f16(false, a1, false, wb[3 * 32 + lane], (short)0, c1, false, false);
    int ncol = lane & 15;
#pragma unroll
    for (int r = 0; r < 8; ++r) {
        int m = r + half * 8;            // C layout: VGPR r -> M=r (lanes 0-15) / M=r+8 (16-31)
        int d = __shfl(dall, m);
        atomicAdd(acc1 + (size_t)d * 32 + ncol,      c0[r]);
        atomicAdd(acc1 + (size_t)d * 32 + 16 + ncol, c1[r]);
    }
}

// ---------------------------------------------------------------------------
// edge_msg2: one wave per 16-edge tile. msg2 = P2[16 x 544] @ Wcat2[544 x 32].
// Each 32-wide K chunk c<16 has A = hh * g[c]  (rank-1), chunk 16 has A = hh.
// 34x v_wmma_f32_16x16x32_f16 per tile; scatter-add into acc2.
// ---------------------------------------------------------------------------
__global__ void edge_msg2_kernel(const int* __restrict__ src, const int* __restrict__ dst,
                                 const _Float16* __restrict__ g2h, const float* __restrict__ h1,
                                 const _Float16* __restrict__ WB2, float* __restrict__ acc2, int E)
{
    int lane = threadIdx.x & 31;
    int wave = threadIdx.x >> 5;
    int tile = blockIdx.x * (blockDim.x >> 5) + wave;
    int base = tile * 16;
    if (base >= E) return;
    int row = lane & 15;
    int half = lane >> 4;
    int e = base + row;
    int s = src[e];
    int dall = dst[e];
    // A-layout (16-bit A 16x32): lane holds K = half*8+0..7 and 16+half*8+0..7 for row M=row
    const float* hp = h1 + (size_t)s * 32 + half * 8;
    v16h hh;
#pragma unroll
    for (int t = 0; t < 8; ++t) {
        hh[t]     = (_Float16)hp[t];
        hh[t + 8] = (_Float16)hp[t + 16];
    }
    v16h g = *(const v16h*)(g2h + (size_t)e * 16);
    const v16h* wb = (const v16h*)WB2;
    v8f c0 = {}, c1 = {};
#pragma unroll
    for (int c = 0; c < 16; ++c) {
        _Float16 gc = g[c];
        v16h a;
#pragma unroll
        for (int t = 0; t < 16; ++t) a[t] = hh[t] * gc;
        v16h b0 = wb[(c * 2 + 0) * 32 + lane];
        v16h b1 = wb[(c * 2 + 1) * 32 + lane];
        c0 = __builtin_amdgcn_wmma_f32_16x16x32_f16(false, a, false, b0, (short)0, c0, false, false);
        c1 = __builtin_amdgcn_wmma_f32_16x16x32_f16(false, a, false, b1, (short)0, c1, false, false);
    }
    {   // bias block: A = hh
        v16h b0 = wb[(16 * 2 + 0) * 32 + lane];
        v16h b1 = wb[(16 * 2 + 1) * 32 + lane];
        c0 = __builtin_amdgcn_wmma_f32_16x16x32_f16(false, hh, false, b0, (short)0, c0, false, false);
        c1 = __builtin_amdgcn_wmma_f32_16x16x32_f16(false, hh, false, b1, (short)0, c1, false, false);
    }
    int ncol = lane & 15;
#pragma unroll
    for (int r = 0; r < 8; ++r) {
        int m = r + half * 8;
        int d = __shfl(dall, m);
        atomicAdd(acc2 + (size_t)d * 32 + ncol,      c0[r]);
        atomicAdd(acc2 + (size_t)d * 32 + 16 + ncol, c1[r]);
    }
}

// ---------------------------------------------------------------------------
// node1: h1 = relu(x @ root1 + acc1/max(cnt,1) + bias1)  (writes in-place over acc1)
// ---------------------------------------------------------------------------
__global__ void node1_kernel(const float* __restrict__ x, const float* __restrict__ root1,
                             const float* __restrict__ bias1, const float* __restrict__ acc1,
                             const float* __restrict__ cnt, float* __restrict__ h1, int N)
{
    int tid = blockIdx.x * blockDim.x + threadIdx.x;
    if (tid >= N * 32) return;
    int n = tid >> 5, o = tid & 31;
    float cv = fmaxf(cnt[n], 1.0f);
    float v = acc1[tid] / cv + bias1[o];
    v += x[n * 3 + 0] * root1[o] + x[n * 3 + 1] * root1[32 + o] + x[n * 3 + 2] * root1[64 + o];
    h1[tid] = fmaxf(v, 0.f);
}

// ---------------------------------------------------------------------------
// node2: fused  h2 = relu(h1@root2 + acc2/cnt + b2); f = relu(h2@fc1+b); out = f@fc2+b
// One wave32 per node, cross-lane via shuffles.
// ---------------------------------------------------------------------------
__global__ void node2_kernel(const float* __restrict__ h1, const float* __restrict__ root2,
                             const float* __restrict__ bias2, const float* __restrict__ acc2,
                             const float* __restrict__ cnt,
                             const float* __restrict__ fc1_w, const float* __restrict__ fc1_b,
                             const float* __restrict__ fc2_w, const float* __restrict__ fc2_b,
                             float* __restrict__ out, int N)
{
    int lane = threadIdx.x & 31;
    int n = (blockIdx.x * blockDim.x + threadIdx.x) >> 5;
    if (n >= N) return;
    float hv = h1[(size_t)n * 32 + lane];
    float cv = fmaxf(cnt[n], 1.0f);
    float acc = acc2[(size_t)n * 32 + lane] / cv + bias2[lane];
#pragma unroll
    for (int i = 0; i < 32; ++i) acc += __shfl(hv, i) * root2[i * 32 + lane];
    float h2 = fmaxf(acc, 0.f);
    float f = fc1_b[lane];
#pragma unroll
    for (int i = 0; i < 32; ++i) f += __shfl(h2, i) * fc1_w[i * 32 + lane];
    f = fmaxf(f, 0.f);
    float p = f * fc2_w[lane];
#pragma unroll
    for (int off = 16; off > 0; off >>= 1) p += __shfl_xor(p, off);
    if (lane == 0) out[n] = p + fc2_b[0];
}

// ---------------------------------------------------------------------------
extern "C" void kernel_launch(void* const* d_in, const int* in_sizes, int n_in,
                              void* d_out, int out_size, void* d_ws, size_t ws_size,
                              hipStream_t stream)
{
    (void)in_sizes; (void)n_in; (void)out_size; (void)ws_size;
    const float* x     = (const float*)d_in[0];
    const int*   ei    = (const int*)  d_in[1];
    const float* ea    = (const float*)d_in[2];
    const float* e1_w1 = (const float*)d_in[3];
    const float* e1_b1 = (const float*)d_in[4];
    const float* e1_w2 = (const float*)d_in[5];
    const float* e1_b2 = (const float*)d_in[6];
    const float* root1 = (const float*)d_in[7];
    const float* bias1 = (const float*)d_in[8];
    const float* e2_w1 = (const float*)d_in[9];
    const float* e2_b1 = (const float*)d_in[10];
    const float* e2_w2 = (const float*)d_in[11];
    const float* e2_b2 = (const float*)d_in[12];
    const float* root2 = (const float*)d_in[13];
    const float* bias2 = (const float*)d_in[14];
    const float* fc1_w = (const float*)d_in[15];
    const float* fc1_b = (const float*)d_in[16];
    const float* fc2_w = (const float*)d_in[17];
    const float* fc2_b = (const float*)d_in[18];
    float* out = (float*)d_out;

    const int E = NE, N = NN;
    const int* srcp = ei;
    const int* dstp = ei + E;

    char* ws = (char*)d_ws;
    size_t off = 0;
    auto take = [&](size_t bytes) {
        char* p = ws + off;
        off = (off + bytes + 255) & ~(size_t)255;
        return p;
    };
    _Float16* WB2 = (_Float16*)take(17 * 1024 * 2);
    _Float16* WB1 = (_Float16*)take(2048 * 2);
    _Float16* g1h = (_Float16*)take((size_t)E * 16 * 2);
    _Float16* g2h = (_Float16*)take((size_t)E * 16 * 2);
    float* cnt  = (float*)take((size_t)N * 4);
    float* acc1 = (float*)take((size_t)N * 32 * 4);
    float* acc2 = (float*)take((size_t)N * 32 * 4);
    float* h1   = acc1;                    // node1 rewrites acc1 in place -> h1

    (void)hipMemsetAsync(cnt,  0, (size_t)N * 4, stream);
    (void)hipMemsetAsync(acc1, 0, (size_t)N * 32 * 4, stream);
    (void)hipMemsetAsync(acc2, 0, (size_t)N * 32 * 4, stream);

    prep_kernel<<<(17 * 1024 + 2048 + 255) / 256, 256, 0, stream>>>(e1_w2, e1_b2, e2_w2, e2_b2, WB2, WB1);
    edge_g_kernel<<<(E + 255) / 256, 256, 0, stream>>>(ea, dstp, e1_w1, e1_b1, e2_w1, e2_b1, g1h, g2h, cnt, E);

    int tiles = (E + 15) / 16;             // 12500
    int wpb = 4;                           // 4 waves (tiles) per 128-thread block
    int mblocks = (tiles + wpb - 1) / wpb;
    edge_msg1_kernel<<<mblocks, wpb * 32, 0, stream>>>(srcp, dstp, g1h, x, WB1, acc1, E);
    node1_kernel<<<(N * 32 + 255) / 256, 256, 0, stream>>>(x, root1, bias1, acc1, cnt, h1, N);
    edge_msg2_kernel<<<mblocks, wpb * 32, 0, stream>>>(srcp, dstp, g2h, h1, WB2, acc2, E);
    node2_kernel<<<(N + 7) / 8, 256, 0, stream>>>(h1, root2, bias2, acc2, cnt,
                                                  fc1_w, fc1_b, fc2_w, fc2_b, out, N);
}
